// MaskedLinear_74423193305702
// MI455X (gfx1250) — matrix operand order repaired and checked by reading, not compiled, not run on hardware
//
#include <hip/hip_runtime.h>
#include <hip/hip_bf16.h>

// Problem sizes (fixed by reference)
#define Bv 32
#define Sv 256
#define Dv 64
#define Hv 512
#define G4 2048        // 4*H
#define SD 16384       // S*D

typedef __attribute__((ext_vector_type(16))) __bf16 v16bf;
typedef __attribute__((ext_vector_type(8)))  float  v8f;

union Afrag { v16bf v; uint2 d[4]; uint4 q[2]; __bf16 h[16]; };

// Native truncation: clang lowers to gfx1250 bf16 convert ops (RNE), far fewer
// VALU than a manual bit-twiddle round.
__device__ __forceinline__ __bf16 f2bf(float f) { return (__bf16)f; }

__device__ __forceinline__ float sigf(float x) { return 1.0f / (1.0f + __expf(-x)); }

// ---- grid-wide software barrier (persistent kernel), deterministic ----
__device__ __forceinline__ void gsync(unsigned* cnt, unsigned* gen,
                                      unsigned nblk, unsigned expect) {
  __threadfence();
  __syncthreads();
  if (threadIdx.x == 0) {
    unsigned prev = __hip_atomic_fetch_add(cnt, 1u, __ATOMIC_ACQ_REL,
                                           __HIP_MEMORY_SCOPE_AGENT);
    if (prev == nblk - 1u) {
      __hip_atomic_store(cnt, 0u, __ATOMIC_RELAXED, __HIP_MEMORY_SCOPE_AGENT);
      __hip_atomic_fetch_add(gen, 1u, __ATOMIC_RELEASE, __HIP_MEMORY_SCOPE_AGENT);
    }
    while (__hip_atomic_load(gen, __ATOMIC_ACQUIRE, __HIP_MEMORY_SCOPE_AGENT) < expect) {
      __builtin_amdgcn_s_sleep(1);
    }
  }
  __syncthreads();
  __threadfence();
}

// ---- Kernel 1: h0 = hidden_0 @ W_hid.T + b_hid ; init LSTM state + barrier ----
__global__ void __launch_bounds__(256) k_init(const float* __restrict__ h0in,
                                              const float* __restrict__ W_hid,
                                              const float* __restrict__ b_hid,
                                              float* out_h0, float* c_st,
                                              __bf16* hbf, unsigned* sync) {
  int tid = blockIdx.x * 256 + threadIdx.x;     // 16384 threads
  if (tid == 0) { sync[0] = 0u; sync[1] = 0u; }
  int b = tid >> 9, hh = tid & 511;
  float acc = b_hid[hh];
  const float* hv = h0in + b * Dv;
  const float* wv = W_hid + hh * Dv;
#pragma unroll 8
  for (int d = 0; d < Dv; ++d) acc += hv[d] * wv[d];
  out_h0[tid] = acc;                            // second output of reference
  c_st[tid]   = acc;                            // c0
  hbf[tid]    = f2bf(acc);                      // h0 (bf16 for WMMA A)
}

// ---- Kernel 2: pack W_hh (row-major [4H,H]) into bf16 WMMA B-fragment tiles ----
// tile (nt,kt): 32 lanes x 16 contiguous-K bf16 each; fragment load = 32B coalesced
__global__ void __launch_bounds__(256) k_pack(const float* __restrict__ W_hh,
                                              __bf16* __restrict__ whhpk) {
  int idx = blockIdx.x * 256 + threadIdx.x;     // 1,048,576
  int tile = idx >> 9, r = idx & 511;
  int lane = r >> 4, j = r & 15;
  int nt = tile >> 4, kt = tile & 15;
  int n = nt * 16 + (lane & 15);
  int k = kt * 32 + ((lane >> 4) << 4) + j;     // lanes>=16 hold K=16..31
  whhpk[idx] = f2bf(W_hh[(size_t)n * Hv + k]);
}

// ---- Kernel 3: G[t] = x[:,t,:] @ W_ih[:, tD:(t+1)D].T  (reads W_ih once) ----
__global__ void __launch_bounds__(256) k_contrib(const float* __restrict__ x,
                                                 const float* __restrict__ W_ih,
                                                 float* __restrict__ G) {
  __shared__ __bf16 xl[32 * 72];                // x_t tile, padded rows
  int t = blockIdx.x >> 2;
  int q = blockIdx.x & 3;
  int tid = threadIdx.x;
  // stage x_t [32x64] as bf16
#pragma unroll
  for (int i = 0; i < 8; ++i) {
    int e = tid * 8 + i; int b = e >> 6, d = e & 63;
    xl[b * 72 + d] = f2bf(x[(size_t)b * SD + t * Dv + d]);
  }
  __syncthreads();

  int w = tid >> 5, lane = tid & 31;
  int mrow  = lane & 15;
  int khalf = (lane >> 4) << 4;                 // A: K base 0/16
  int bhalf = (lane >> 4) << 4;                 // B: K base 0/16
  int mbase = (lane >> 4) << 3;                 // C/D: row base 0/8

  for (int j = 0; j < 4; ++j) {
    int nt = q * 32 + w * 4 + j;                // 0..127
    int n  = nt * 16 + mrow;
    v8f acc0 = {0,0,0,0,0,0,0,0};
    v8f acc1 = {0,0,0,0,0,0,0,0};
#pragma unroll
    for (int kt = 0; kt < 2; ++kt) {
      // B fragment: 16 consecutive f32 of W_ih row n (contiguous in K)
      const float* bp = W_ih + (size_t)n * SD + t * Dv + kt * 32 + bhalf;
      float4 f0 = ((const float4*)bp)[0];
      float4 f1 = ((const float4*)bp)[1];
      float4 f2 = ((const float4*)bp)[2];
      float4 f3 = ((const float4*)bp)[3];
      Afrag bb;
      bb.h[0]=f2bf(f0.x); bb.h[1]=f2bf(f0.y); bb.h[2]=f2bf(f0.z); bb.h[3]=f2bf(f0.w);
      bb.h[4]=f2bf(f1.x); bb.h[5]=f2bf(f1.y); bb.h[6]=f2bf(f1.z); bb.h[7]=f2bf(f1.w);
      bb.h[8]=f2bf(f2.x); bb.h[9]=f2bf(f2.y); bb.h[10]=f2bf(f2.z); bb.h[11]=f2bf(f2.w);
      bb.h[12]=f2bf(f3.x); bb.h[13]=f2bf(f3.y); bb.h[14]=f2bf(f3.z); bb.h[15]=f2bf(f3.w);
      // A fragments from LDS: one contiguous 16-elem K run per lane
      int kb = kt * 32 + khalf;
      Afrag a0, a1;
      const uint2* p0 = (const uint2*)(&xl[mrow * 72 + kb]);
      a0.d[0] = p0[0]; a0.d[1] = p0[1]; a0.d[2] = p0[2]; a0.d[3] = p0[3];
      const uint2* p1 = (const uint2*)(&xl[(16 + mrow) * 72 + kb]);
      a1.d[0] = p1[0]; a1.d[1] = p1[1]; a1.d[2] = p1[2]; a1.d[3] = p1[3];
      acc0 = __builtin_amdgcn_wmma_f32_16x16x32_bf16(false, a0.v, false, bb.v,
                                                     (short)0, acc0, false, false);
      acc1 = __builtin_amdgcn_wmma_f32_16x16x32_bf16(false, a1.v, false, bb.v,
                                                     (short)0, acc1, false, false);
    }
    float* gp = G + (size_t)t * (Bv * G4);
    int N = nt * 16 + (lane & 15);
#pragma unroll
    for (int r = 0; r < 8; ++r) {
      int M0 = mbase + r;
      gp[M0 * G4 + N]        = acc0[r];
      gp[(16 + M0) * G4 + N] = acc1[r];
    }
  }
}

// ---- Kernel 4: in-place prefix scan over t, fold in (b_ih + b_hh) ----
__global__ void __launch_bounds__(256) k_scan(float* __restrict__ G,
                                              const float* __restrict__ b_ih,
                                              const float* __restrict__ b_hh) {
  int idx = blockIdx.x * 256 + threadIdx.x;     // 65536 = B*4H
  int n = idx & (G4 - 1);
  float bias = b_ih[n] + b_hh[n];
  float run = 0.0f;
  for (int t = 0; t < Sv; ++t) {
    size_t a = (size_t)t * (Bv * G4) + idx;
    run += G[a];
    G[a] = run + bias;
  }
}

// ---- Kernel 5: persistent LSTM recurrence (16 WGs, 128 waves, 1 N-tile/wave) ----
__global__ void __launch_bounds__(256) k_recur(const float* __restrict__ P,
                                               const __bf16* __restrict__ whhpk,
                                               float* __restrict__ gates,
                                               float* __restrict__ c_st,
                                               __bf16* __restrict__ hbf,
                                               float* __restrict__ out,
                                               unsigned* sync) {
  int tid = threadIdx.x;
  int w = tid >> 5, lane = tid & 31;
  int nt = blockIdx.x * 8 + w;                  // 0..127
  int mrow  = lane & 15;
  int khalf = (lane >> 4) << 4;
  int mbase = (lane >> 4) << 3;
  int N = nt * 16 + (lane & 15);
  int gtid = blockIdx.x * 256 + tid;            // 0..4095 for elementwise phase

  for (int t = 0; t < Sv; ++t) {
    // ---- phase A: gates = P[t] + h @ W_hh.T (WMMA over K=512) ----
    v8f acc0 = {0,0,0,0,0,0,0,0};
    v8f acc1 = {0,0,0,0,0,0,0,0};
#pragma unroll 4
    for (int kt = 0; kt < 16; ++kt) {
      v16bf bb = *(const v16bf*)(whhpk + (((size_t)nt * 16 + kt) << 9) + (lane << 4));
      int kb = kt * 32 + khalf;
      Afrag a0, a1;
      const uint4* p0 = (const uint4*)(hbf + mrow * Hv + kb);
      a0.q[0] = p0[0]; a0.q[1] = p0[1];
      const uint4* p1 = (const uint4*)(hbf + (16 + mrow) * Hv + kb);
      a1.q[0] = p1[0]; a1.q[1] = p1[1];
      acc0 = __builtin_amdgcn_wmma_f32_16x16x32_bf16(false, a0.v, false, bb,
                                                     (short)0, acc0, false, false);
      acc1 = __builtin_amdgcn_wmma_f32_16x16x32_bf16(false, a1.v, false, bb,
                                                     (short)0, acc1, false, false);
    }
    const float* Pt = P + (size_t)t * (Bv * G4);
#pragma unroll
    for (int r = 0; r < 8; ++r) {
      int M0 = mbase + r;
      gates[M0 * G4 + N]        = acc0[r] + Pt[M0 * G4 + N];
      gates[(16 + M0) * G4 + N] = acc1[r] + Pt[(16 + M0) * G4 + N];
    }
    gsync(sync, sync + 1, 16u, (unsigned)(2 * t + 1));

    // Prefetch next step's P slice (256 KB, one prefetch per 64B across 4096
    // threads) so its L2/HBM latency overlaps the next GEMM phase.
    // At t==255 this lands in the adjacent `gates` scratch (mapped, harmless).
    __builtin_prefetch(&P[(size_t)(t + 1) * (Bv * G4) + ((size_t)gtid << 4)], 0, 3);

    // ---- phase B: elementwise LSTM cell, write relu(h) to out ----
#pragma unroll
    for (int kk = 0; kk < 4; ++kk) {
      int e = gtid + (kk << 12);                // 0..16383 == b*512+hh
      int b = e >> 9, hh = e & 511;
      int gb = b * G4 + hh;
      float ig = sigf(gates[gb]);
      float fg = sigf(gates[gb + 512]);
      float gg = tanhf(gates[gb + 1024]);
      float og = sigf(gates[gb + 1536]);
      float c = fg * c_st[e] + ig * gg;
      float h = og * tanhf(c);
      c_st[e] = c;
      hbf[e]  = f2bf(h);
      out[(size_t)b * (Sv * Hv) + t * Hv + hh] = fmaxf(h, 0.0f);
    }
    gsync(sync, sync + 1, 16u, (unsigned)(2 * t + 2));
  }
}

extern "C" void kernel_launch(void* const* d_in, const int* in_sizes, int n_in,
                              void* d_out, int out_size, void* d_ws, size_t ws_size,
                              hipStream_t stream) {
  (void)in_sizes; (void)n_in; (void)out_size; (void)ws_size;
  const float* x     = (const float*)d_in[0];
  const float* h0in  = (const float*)d_in[1];
  const float* W_hid = (const float*)d_in[2];
  const float* b_hid = (const float*)d_in[3];
  const float* W_ih  = (const float*)d_in[4];
  const float* W_hh  = (const float*)d_in[5];
  const float* b_ih  = (const float*)d_in[6];
  const float* b_hh  = (const float*)d_in[7];

  float* out    = (float*)d_out;
  float* out_h0 = out + (size_t)Bv * Sv * Hv;   // second reference output

  char* ws = (char*)d_ws;
  float*    P     = (float*)(ws);               // 64 MB: S x B x 4H prefix sums
  float*    gates = (float*)(ws + 67108864);    // 256 KB: B x 4H
  float*    c_st  = (float*)(ws + 67371008);    // 64 KB: B x H
  __bf16*   hbf   = (__bf16*)(ws + 67436544);   // 32 KB: B x H bf16
  __bf16*   whhpk = (__bf16*)(ws + 67469312);   // 2 MB: packed W_hh tiles
  unsigned* syncv = (unsigned*)(ws + 69566464); // barrier counter + generation

  k_init   <<<  64, 256, 0, stream>>>(h0in, W_hid, b_hid, out_h0, c_st, hbf, syncv);
  k_pack   <<<4096, 256, 0, stream>>>(W_hh, whhpk);
  k_contrib<<<1024, 256, 0, stream>>>(x, W_ih, P);
  k_scan   <<< 256, 256, 0, stream>>>(P, b_ih, b_hh);
  k_recur  <<<  16, 256, 0, stream>>>(P, whhpk, gates, c_st, hbf, out, syncv);
}
